// Linear3Bit_47656957116916
// MI455X (gfx1250) — compile-verified, matrix operand order checked by reading, not run on hardware
//
#include <hip/hip_runtime.h>

typedef __attribute__((ext_vector_type(16))) __bf16 v16bf;
typedef __attribute__((ext_vector_type(8)))  __bf16 v8bf;
typedef __attribute__((ext_vector_type(8)))  float  v8f;

namespace {
constexpr int IN_F  = 4096;
constexpr int OUT_F = 11008;
constexpr int NUM_GROUPS = OUT_F * IN_F / 16;   // 2,818,048

// ---------------- fp32 -> packed 2x bf16 (RNE) ----------------
__device__ __forceinline__ unsigned int cvt2(float lo, float hi) {
#if __has_builtin(__builtin_amdgcn_cvt_pk_bf16_f32)
    typedef __attribute__((ext_vector_type(2))) __bf16 v2bf;
    v2bf r = __builtin_amdgcn_cvt_pk_bf16_f32(lo, hi);
    return __builtin_bit_cast(unsigned int, r);
#else
    unsigned int ulo = __builtin_bit_cast(unsigned int, lo);
    unsigned int uhi = __builtin_bit_cast(unsigned int, hi);
    ulo += 0x7FFFu + ((ulo >> 16) & 1u);
    uhi += 0x7FFFu + ((uhi >> 16) & 1u);
    return (ulo >> 16) | (uhi & 0xFFFF0000u);
#endif
}

__device__ __forceinline__ v16bf cat16(v8bf lo, v8bf hi) {
    return __builtin_shufflevector(lo, hi, 0, 1, 2, 3, 4, 5, 6, 7,
                                           8, 9, 10, 11, 12, 13, 14, 15);
}

// -------- gfx1250 async global->LDS copy (ASYNCcnt-tracked, no VGPR data) --------
template <int OFF>
__device__ __forceinline__ void async_b128(unsigned lds_addr, unsigned voff,
                                           const void* sbase) {
    asm volatile("global_load_async_to_lds_b128 %0, %1, %2 offset:%3"
                 :: "v"(lds_addr), "v"(voff), "s"(sbase), "n"(OFF)
                 : "memory");
}

template <int N>
__device__ __forceinline__ void wait_async() {
#if __has_builtin(__builtin_amdgcn_s_wait_asynccnt)
    __builtin_amdgcn_s_wait_asynccnt(N);
#else
    asm volatile("s_wait_asynccnt %0" :: "n"(N) : "memory");
#endif
}
} // namespace

// =====================================================================
// Prep kernel 1: x fp32 -> bf16 (8 elements / thread)
// =====================================================================
__global__ void __launch_bounds__(256)
prep_x_kernel(const float* __restrict__ x, unsigned short* __restrict__ xbf)
{
    const size_t idx = ((size_t)blockIdx.x * 256 + threadIdx.x) * 8;
    float4 a = *reinterpret_cast<const float4*>(x + idx);
    float4 b = *reinterpret_cast<const float4*>(x + idx + 4);
    uint4 p = make_uint4(cvt2(a.x, a.y), cvt2(a.z, a.w),
                         cvt2(b.x, b.y), cvt2(b.z, b.w));
    *reinterpret_cast<uint4*>(xbf + idx) = p;
}

// =====================================================================
// Prep kernel 2: 3-bit groups -> bf16 weights [OUT_F, IN_F] row-major
// =====================================================================
__global__ void __launch_bounds__(256)
prep_w_kernel(const int* __restrict__ wq, const float* __restrict__ wn,
              unsigned short* __restrict__ wbf)
{
    const size_t g = (size_t)blockIdx.x * 256 + threadIdx.x;
    const int* qp = wq + g * 8;
    int4 q0 = *reinterpret_cast<const int4*>(qp);
    int4 q1 = *reinterpret_cast<const int4*>(qp + 4);
    const float nrm   = wn[g];
    const float scale = nrm * (2.0f / 7.0f);
    const float wb    = -nrm;
    const int qv[8] = {q0.x, q0.y, q0.z, q0.w, q1.x, q1.y, q1.z, q1.w};
    unsigned int pk[8];
    #pragma unroll
    for (int p = 0; p < 8; ++p) {
        const float w0 = fmaf((float)(qv[p] & 7),        scale, wb);
        const float w1 = fmaf((float)((qv[p] >> 3) & 7), scale, wb);
        pk[p] = cvt2(w0, w1);
    }
    unsigned short* op = wbf + g * 16;
    *reinterpret_cast<uint4*>(op)     = make_uint4(pk[0], pk[1], pk[2], pk[3]);
    *reinterpret_cast<uint4*>(op + 8) = make_uint4(pk[4], pk[5], pk[6], pk[7]);
}

// =====================================================================
// Main GEMM: bf16 A [M,K] x bf16 B [N,K] -> f32, WMMA 16x16x32.
// 256x128 C tile / 256 threads; 8 waves as 4x2, each computing 64x64
// (acc[4][4], 16 WMMA per 32-K slab -> 8KB LDS per 131K MACs).
// Double-buffered LDS fed by global_load_async_to_lds_b128.
// =====================================================================
namespace {
constexpr int BM = 256, BN = 128, BK = 32;
constexpr int LDS_K = BK + 8;               // 40 bf16 = 80B stride, 16B aligned, skewed
constexpr int KT   = IN_F / BK;             // 128 slabs
constexpr int ABUF = BM * LDS_K;            // 10240 ushorts = 20 KB
constexpr int BBUF = BN * LDS_K;            //  5120 ushorts = 10 KB
}

__global__ void __launch_bounds__(256, 1)
gemm_bf16_wmma_kernel(const unsigned short* __restrict__ xb,
                      const unsigned short* __restrict__ wb,
                      const float* __restrict__ bias,
                      float* __restrict__ out)
{
    __shared__ unsigned short As[2][ABUF];   // 40 KB
    __shared__ unsigned short Bs[2][BBUF];   // 20 KB  (61,440 B total)

    const int tid  = threadIdx.x;
    const int lane = tid & 31;
    const int wave = tid >> 5;
    const int wm   = wave >> 1;      // 0..3 : 64-row slice of C tile
    const int wnv  = wave & 1;       // 0..1 : 64-col slice of C tile
    const int half = lane >> 4;
    const int r16  = lane & 15;

    const int n0 = blockIdx.x * BN;
    const int m0 = blockIdx.y * BM;

    // staging roles: A = 1 row/thread (64B, 4 async b128);
    //                B = 2 threads/row (32B each, 2 async b128)
    const int brow = tid >> 1;
    const int boff = (tid & 1) << 4;                 // 0 or 16 elements

    const unsigned voffA0 = ((unsigned)(m0 + tid)  * IN_F) * 2u;
    const unsigned voffB0 = ((unsigned)(n0 + brow) * IN_F + (unsigned)boff) * 2u;

    unsigned ldsA[2], ldsB[2];
    #pragma unroll
    for (int b = 0; b < 2; ++b) {
        ldsA[b] = (unsigned)(uintptr_t)&As[b][tid * LDS_K];
        ldsB[b] = (unsigned)(uintptr_t)&Bs[b][brow * LDS_K + boff];
    }

    const v8f zacc = {0.f, 0.f, 0.f, 0.f, 0.f, 0.f, 0.f, 0.f};
    v8f acc[4][4];
    #pragma unroll
    for (int i = 0; i < 4; ++i)
        #pragma unroll
        for (int j = 0; j < 4; ++j) acc[i][j] = zacc;

    // prologue: slab 0 into buffer 0
    async_b128<0 >(ldsA[0], voffA0, xb);
    async_b128<16>(ldsA[0], voffA0, xb);
    async_b128<32>(ldsA[0], voffA0, xb);
    async_b128<48>(ldsA[0], voffA0, xb);
    async_b128<0 >(ldsB[0], voffB0, wb);
    async_b128<16>(ldsB[0], voffB0, wb);
    wait_async<0>();
    __syncthreads();

    for (int kt = 0; kt < KT; ++kt) {
        const int cur = kt & 1;
        // issue next slab's async copies; they land during the 16 WMMAs below
        if (kt + 1 < KT) {
            const int nb = cur ^ 1;
            const unsigned va = voffA0 + (unsigned)(kt + 1) * (BK * 2);
            const unsigned vb = voffB0 + (unsigned)(kt + 1) * (BK * 2);
            async_b128<0 >(ldsA[nb], va, xb);
            async_b128<16>(ldsA[nb], va, xb);
            async_b128<32>(ldsA[nb], va, xb);
            async_b128<48>(ldsA[nb], va, xb);
            async_b128<0 >(ldsB[nb], vb, wb);
            async_b128<16>(ldsB[nb], vb, wb);
        }

        const unsigned short* Ac = &As[cur][0];
        const unsigned short* Bc = &Bs[cur][0];

        v16bf bfr[4];
        #pragma unroll
        for (int j = 0; j < 4; ++j) {
            const unsigned short* p = &Bc[(wnv * 64 + j * 16 + r16) * LDS_K + half * 8];
            bfr[j] = cat16(*reinterpret_cast<const v8bf*>(p),
                           *reinterpret_cast<const v8bf*>(p + 16));
        }
        #pragma unroll
        for (int i = 0; i < 4; ++i) {
            const unsigned short* p = &Ac[(wm * 64 + i * 16 + r16) * LDS_K + half * 8];
            v16bf af = cat16(*reinterpret_cast<const v8bf*>(p),
                             *reinterpret_cast<const v8bf*>(p + 16));
            #pragma unroll
            for (int j = 0; j < 4; ++j)
                acc[i][j] = __builtin_amdgcn_wmma_f32_16x16x32_bf16(
                    false, af, false, bfr[j], (short)0, acc[i][j], false, false);
        }

        wait_async<0>();          // own async copies for slab kt+1 landed
        __syncthreads();          // everyone's landed; reads of buf[cur] done
    }

    float bv[4];
    #pragma unroll
    for (int j = 0; j < 4; ++j)
        bv[j] = bias[n0 + wnv * 64 + j * 16 + r16];
    #pragma unroll
    for (int i = 0; i < 4; ++i) {
        const int gmb = m0 + wm * 64 + i * 16 + half * 8;
        #pragma unroll
        for (int j = 0; j < 4; ++j) {
            const int gn = n0 + wnv * 64 + j * 16 + r16;
            #pragma unroll
            for (int r = 0; r < 8; ++r)
                out[(size_t)(gmb + r) * OUT_F + gn] = acc[i][j][r] + bv[j];
        }
    }
}

// =====================================================================
// Fallback: fused dequant+GEMM, double-buffered LDS, one barrier/slab
// (used only if workspace is too small). 128x128 tile, 8 waves as 2x4.
// =====================================================================
namespace {
constexpr int FBM = 128, FBN = 128;
constexpr int FLD = BK + 8;                 // 40
constexpr int FKT = IN_F / BK;              // 128
}

__global__ void __launch_bounds__(256, 1)
linear3bit_fused_kernel(const float* __restrict__ x,
                        const int*   __restrict__ wq,
                        const float* __restrict__ wn,
                        const float* __restrict__ bias,
                        float* __restrict__ out)
{
    __shared__ unsigned short As[2][FBM * FLD];
    __shared__ unsigned short Bs[2][FBN * FLD];

    const int tid  = threadIdx.x;
    const int lane = tid & 31;
    const int wave = tid >> 5;
    const int wm   = wave >> 2;
    const int wnv  = wave & 3;
    const int half = lane >> 4;
    const int r16  = lane & 15;

    const int n0 = blockIdx.x * FBN;
    const int m0 = blockIdx.y * FBM;

    const int arow = tid >> 1;
    const int acol = (tid & 1) << 4;
    const int bgi  = tid & 1;

    const float* xptr = x + (size_t)(m0 + arow) * IN_F + acol;
    const long   g0   = (long)(n0 + arow) * (IN_F / 16) + bgi;

    const v8f zacc = {0.f, 0.f, 0.f, 0.f, 0.f, 0.f, 0.f, 0.f};
    v8f acc[4][2];
    #pragma unroll
    for (int i = 0; i < 4; ++i)
        #pragma unroll
        for (int j = 0; j < 2; ++j) acc[i][j] = zacc;

    float4 ax[4];
    int4   q0, q1;
    float  nrm;

    auto load_slab = [&](int kt) {
        const int kb = kt * BK;
        #pragma unroll
        for (int c = 0; c < 4; ++c)
            ax[c] = *reinterpret_cast<const float4*>(xptr + kb + c * 4);
        const long g = g0 + (long)(kb >> 4);
        const int* qp = wq + (size_t)g * 8;
        q0 = *reinterpret_cast<const int4*>(qp);
        q1 = *reinterpret_cast<const int4*>(qp + 4);
        nrm = wn[g];
    };
    auto store_slab = [&](int b) {
        uint4 p0 = make_uint4(cvt2(ax[0].x, ax[0].y), cvt2(ax[0].z, ax[0].w),
                              cvt2(ax[1].x, ax[1].y), cvt2(ax[1].z, ax[1].w));
        uint4 p1 = make_uint4(cvt2(ax[2].x, ax[2].y), cvt2(ax[2].z, ax[2].w),
                              cvt2(ax[3].x, ax[3].y), cvt2(ax[3].z, ax[3].w));
        unsigned short* ap = &As[b][arow * FLD + acol];
        *reinterpret_cast<uint4*>(ap)     = p0;
        *reinterpret_cast<uint4*>(ap + 8) = p1;

        const float scale = nrm * (2.0f / 7.0f);
        const float wbv   = -nrm;
        const int qv[8] = {q0.x, q0.y, q0.z, q0.w, q1.x, q1.y, q1.z, q1.w};
        unsigned int pk[8];
        #pragma unroll
        for (int p = 0; p < 8; ++p) {
            const float w0 = fmaf((float)(qv[p] & 7),        scale, wbv);
            const float w1 = fmaf((float)((qv[p] >> 3) & 7), scale, wbv);
            pk[p] = cvt2(w0, w1);
        }
        unsigned short* bp = &Bs[b][arow * FLD + bgi * 16];
        *reinterpret_cast<uint4*>(bp)     = make_uint4(pk[0], pk[1], pk[2], pk[3]);
        *reinterpret_cast<uint4*>(bp + 8) = make_uint4(pk[4], pk[5], pk[6], pk[7]);
    };

    load_slab(0);
    store_slab(0);
    __syncthreads();

    for (int kt = 0; kt < FKT; ++kt) {
        const bool more = (kt + 1 < FKT);
        if (more) load_slab(kt + 1);

        const int cur = kt & 1;
        const unsigned short* Ac = &As[cur][0];
        const unsigned short* Bc = &Bs[cur][0];
        v16bf bfr[2];
        #pragma unroll
        for (int j = 0; j < 2; ++j) {
            const unsigned short* p = &Bc[(wnv * 32 + j * 16 + r16) * FLD + half * 8];
            bfr[j] = cat16(*reinterpret_cast<const v8bf*>(p),
                           *reinterpret_cast<const v8bf*>(p + 16));
        }
        #pragma unroll
        for (int i = 0; i < 4; ++i) {
            const unsigned short* p = &Ac[(wm * 64 + i * 16 + r16) * FLD + half * 8];
            v16bf af = cat16(*reinterpret_cast<const v8bf*>(p),
                             *reinterpret_cast<const v8bf*>(p + 16));
            acc[i][0] = __builtin_amdgcn_wmma_f32_16x16x32_bf16(
                false, af, false, bfr[0], (short)0, acc[i][0], false, false);
            acc[i][1] = __builtin_amdgcn_wmma_f32_16x16x32_bf16(
                false, af, false, bfr[1], (short)0, acc[i][1], false, false);
        }

        if (more) store_slab((kt + 1) & 1);
        __syncthreads();
    }

    const float bv0 = bias[n0 + wnv * 32 + r16];
    const float bv1 = bias[n0 + wnv * 32 + 16 + r16];
    #pragma unroll
    for (int i = 0; i < 4; ++i) {
        const int gmb = m0 + wm * 64 + i * 16 + half * 8;
        #pragma unroll
        for (int j = 0; j < 2; ++j) {
            const int   gn = n0 + wnv * 32 + j * 16 + r16;
            const float bj = j ? bv1 : bv0;
            #pragma unroll
            for (int r = 0; r < 8; ++r)
                out[(size_t)(gmb + r) * OUT_F + gn] = acc[i][j][r] + bj;
        }
    }
}

// =====================================================================
extern "C" void kernel_launch(void* const* d_in, const int* in_sizes, int n_in,
                              void* d_out, int out_size, void* d_ws, size_t ws_size,
                              hipStream_t stream) {
    const float* x    = (const float*)d_in[0];
    const int*   wq   = (const int*)d_in[1];
    const float* wn   = (const float*)d_in[2];
    const float* bias = (const float*)d_in[3];
    float*       out  = (float*)d_out;
    const int M = in_sizes[0] / IN_F;            // 8192

    const size_t xbytes = (size_t)M * IN_F * sizeof(unsigned short);
    const size_t wbytes = (size_t)OUT_F * IN_F * sizeof(unsigned short);

    if (d_ws != nullptr && ws_size >= xbytes + wbytes) {
        unsigned short* xbf = (unsigned short*)d_ws;
        unsigned short* wbf = (unsigned short*)((char*)d_ws + xbytes);
        prep_x_kernel<<<(int)(((size_t)M * IN_F) / (256 * 8)), dim3(256), 0, stream>>>(x, xbf);
        prep_w_kernel<<<NUM_GROUPS / 256, dim3(256), 0, stream>>>(wq, wn, wbf);
        dim3 grid(OUT_F / BN, M / BM);           // (86, 32)
        gemm_bf16_wmma_kernel<<<grid, dim3(256), 0, stream>>>(xbf, wbf, bias, out);
    } else {
        dim3 grid(OUT_F / FBN, M / FBM);         // (86, 64)
        linear3bit_fused_kernel<<<grid, dim3(256), 0, stream>>>(x, wq, wn, bias, out);
    }
    (void)n_in; (void)out_size;
}
